// PatchSelfConv_592705487160
// MI455X (gfx1250) — matrix operand order repaired and checked by reading, not compiled, not run on hardware
//
#include <hip/hip_runtime.h>

// ---------------------------------------------------------------------------
// PatchSelfConv on MI455X (gfx1250, wave32, WMMA)
//
// out[b,n,y,x] = sum_{c,i,j} img[b,c,y+i,x+j] * img[b,c,(n/14)*16+i,(n%14)*16+j]
//
// Per-batch GEMM  C[196 x 43681] = W[196 x 768] * im2col[768 x 43681]
// via V_WMMA_F32_16X16X32_BF16 with 2xBF16 split of fp32 (hi*hi + hi*lo +
// lo*hi), fp32 accumulation. im2col is never materialized: each workgroup
// stages one 3x16x224 image-row slab + one 16x768 patch tile in LDS.
//
// LDS layouts are arranged so inner-loop fragment loads are wide & aligned:
//   A: fragment-major  -> 32B-contiguous per lane  (ds_load_b128 x2 / half)
//   I: base + shifted-by-1 copy -> 4B-aligned dwords (ds_load_b32 x8 / half)
// ---------------------------------------------------------------------------

typedef __attribute__((ext_vector_type(16))) __bf16        v16bf;
typedef __attribute__((ext_vector_type(8)))  float         v8f;
typedef __attribute__((ext_vector_type(8)))  unsigned int  v8u;

#define IN_C     3
#define IMG      224
#define PATCHSZ  16
#define NPR      14          // patches per image row
#define NPATCH   196
#define OUTW     209         // 224 - 16 + 1
#define K_TOT    768         // 3*16*16
#define KSTEPS   24          // 768 / 32
#define PT_TILES 13          // ceil(196/16)
#define XT       14          // ceil(209/16) x-tiles per output row
#define THREADS  128
#define NWAVES   4
#define TPW      4           // x-tiles per wave (4 waves * 4 >= 14)

#define A_ELEMS  (K_TOT * 16)            // 12288 bf16 per copy (24 KB)
#define I_ELEMS  (IN_C * PATCHSZ * IMG)  // 10752 bf16 per copy (21 KB)
// A_hi, A_lo (frag-major) + I_hi, I_lo + shifted copies I_hi_s, I_lo_s
#define LDS_BYTES ((2 * A_ELEMS + 4 * I_ELEMS) * 2)   // 135168 B = 132 KB

__device__ __forceinline__ void bf16_split(float v, __bf16& h, __bf16& l) {
    __bf16 hb = (__bf16)v;            // RNE to bf16
    float  hf = (float)hb;
    h = hb;
    l = (__bf16)(v - hf);             // residual: ~8 more mantissa bits
}

__global__ void __launch_bounds__(THREADS)
patch_self_conv_wmma(const float* __restrict__ x, float* __restrict__ out) {
    extern __shared__ __align__(16) __bf16 smem[];
    __bf16* Ahi  = smem;                 // [KSTEPS][32 lanes][16 e]  frag-major
    __bf16* Alo  = Ahi  + A_ELEMS;
    __bf16* Ihi  = Alo  + A_ELEMS;       // [3*16][224]
    __bf16* Ilo  = Ihi  + I_ELEMS;
    __bf16* IhiS = Ilo  + I_ELEMS;       // IhiS[i] == Ihi[i+1]
    __bf16* IloS = IhiS + I_ELEMS;

    const int y    = blockIdx.x;         // output row      0..208
    const int pt   = blockIdx.y;         // 16-patch tile   0..12
    const int b    = blockIdx.z;         // batch           0..7
    const int tid  = threadIdx.x;
    const int lane = tid & 31;
    const int wave = tid >> 5;
    const int hi   = lane >> 4;          // lane half (WMMA layout)
    const int mn   = lane & 15;          // M (C rows) or N (B/C cols)

    const float* imgB = x + (size_t)b * IN_C * IMG * IMG;

    // ---- Stage 1: image rows y..y+15 (all ch) -> LDS bf16 hi/lo + shifted --
    for (int idx = tid; idx < I_ELEMS; idx += THREADS) {
        int row = idx / IMG;             // c*16 + i
        int xx  = idx - row * IMG;
        int c   = row >> 4;
        int i   = row & 15;
        float v = imgB[(c * IMG + (y + i)) * IMG + xx];
        __bf16 h, l;
        bf16_split(v, h, l);
        Ihi[idx] = h;  Ilo[idx] = l;
        if (idx > 0) { IhiS[idx - 1] = h;  IloS[idx - 1] = l; }
        else         { IhiS[I_ELEMS - 1] = (__bf16)0.f;
                       IloS[I_ELEMS - 1] = (__bf16)0.f; }
    }

    // ---- Stage 2: patch-weight tile -> LDS in WMMA fragment-major order ----
    // memory element ((ks*32 + frag_lane)*16 + e) holds
    //   A[k][m],  k = ks*32 + e + 16*(frag_lane>>4),  m = frag_lane&15
    for (int idx = tid; idx < A_ELEMS; idx += THREADS) {
        int e   = idx & 15;
        int fl  = (idx >> 4) & 31;       // fragment lane
        int ks  = idx >> 9;
        int k   = ks * 32 + e + ((fl >> 4) << 4);
        int m   = fl & 15;
        int pn  = pt * 16 + m;
        float v = 0.0f;
        if (pn < NPATCH) {
            int c  = k >> 8;
            int i  = (k >> 4) & 15;
            int jj = k & 15;
            int py = pn / NPR;
            int px = pn - py * NPR;
            v = imgB[(c * IMG + (py * PATCHSZ + i)) * IMG + (px * PATCHSZ + jj)];
        }
        bf16_split(v, Ahi[idx], Alo[idx]);
    }
    __syncthreads();

    // dword views for wide aligned fragment loads
    const unsigned int* UAhi  = (const unsigned int*)Ahi;
    const unsigned int* UAlo  = (const unsigned int*)Alo;
    const unsigned int* UIhi  = (const unsigned int*)Ihi;
    const unsigned int* UIlo  = (const unsigned int*)Ilo;
    const unsigned int* UIhiS = (const unsigned int*)IhiS;
    const unsigned int* UIloS = (const unsigned int*)IloS;

    // ---- Stage 3: GEMM sweep along the output row --------------------------
    v8f acc[TPW];
#pragma unroll
    for (int j = 0; j < TPW; ++j) acc[j] = (v8f){0.f,0.f,0.f,0.f,0.f,0.f,0.f,0.f};

    for (int ks = 0; ks < KSTEPS; ++ks) {
        // A fragments: 8 consecutive dwords at a 32B-aligned address
        const int aw = (ks * 32 + lane) * 8;       // dword index
        v8u awh, awl;
#pragma unroll
        for (int q = 0; q < 8; ++q) { awh[q] = UAhi[aw + q]; awl[q] = UAlo[aw + q]; }
        const v16bf ahi = __builtin_bit_cast(v16bf, awh);
        const v16bf alo = __builtin_bit_cast(v16bf, awl);

        // B row base: lane half covers K = ks*32 + hi*16 .. +15 -> one fixed
        // (c,i) row, jj = e. Elements are contiguous in the row slab.
        const int k0   = ks * 32 + hi * 16;
        const int rowb = (((k0 >> 8) << 4) + ((k0 >> 4) & 15)) * IMG;

        // odd lanes use the shifted copy so the dword base is 4B-aligned
        const unsigned int* srcH = (mn & 1) ? UIhiS : UIhi;
        const unsigned int* srcL = (mn & 1) ? UIloS : UIlo;

#pragma unroll
        for (int j = 0; j < TPW; ++j) {
            int t = wave + NWAVES * j;             // wave-uniform guard
            if (t < XT) {
                const int w = (rowb + t * 16 + (mn & ~1)) >> 1;  // dword index
                v8u bwh, bwl;
#pragma unroll
                for (int q = 0; q < 8; ++q) { bwh[q] = srcH[w + q]; bwl[q] = srcL[w + q]; }
                const v16bf bhi = __builtin_bit_cast(v16bf, bwh);
                const v16bf blo = __builtin_bit_cast(v16bf, bwl);

                // fp32 ~= hi*hi + hi*lo + lo*hi (f32 accumulate)
                acc[j] = __builtin_amdgcn_wmma_f32_16x16x32_bf16(
                             false, ahi, false, bhi, (short)0, acc[j], false, false);
                acc[j] = __builtin_amdgcn_wmma_f32_16x16x32_bf16(
                             false, ahi, false, blo, (short)0, acc[j], false, false);
                acc[j] = __builtin_amdgcn_wmma_f32_16x16x32_bf16(
                             false, alo, false, bhi, (short)0, acc[j], false, false);
            }
        }
    }

    // ---- Stage 4: epilogue. C layout: M = v + 8*hi, N = lane&15 ------------
#pragma unroll
    for (int j = 0; j < TPW; ++j) {
        int t = wave + NWAVES * j;
        if (t < XT) {
            int xcol = t * 16 + mn;
            if (xcol < OUTW) {
#pragma unroll
                for (int v = 0; v < 8; ++v) {
                    int pn = pt * 16 + hi * 8 + v;
                    if (pn < NPATCH) {
                        out[(((size_t)b * NPATCH + pn) * OUTW + y) * OUTW + xcol] =
                            acc[j][v];
                    }
                }
            }
        }
    }
}

extern "C" void kernel_launch(void* const* d_in, const int* in_sizes, int n_in,
                              void* d_out, int out_size, void* d_ws, size_t ws_size,
                              hipStream_t stream) {
    (void)in_sizes; (void)n_in; (void)out_size; (void)d_ws; (void)ws_size;
    const float* x = (const float*)d_in[0];
    float* out     = (float*)d_out;

    dim3 grid(OUTW, PT_TILES, 8);   // (y row, patch tile, batch)
    dim3 block(THREADS);            // 4 waves (wave32)
    patch_self_conv_wmma<<<grid, block, LDS_BYTES, stream>>>(x, out);
}
